// MHSA_80736795231110
// MI455X (gfx1250) — matrix operand order repaired and checked by reading, not compiled, not run on hardware
//
#include <hip/hip_runtime.h>
#include <hip/hip_bf16.h>

typedef __bf16 bf16_t;
typedef __attribute__((ext_vector_type(16))) __bf16 v16bf;
typedef __attribute__((ext_vector_type(8)))  float  v8f;

#define B_SZ 4
#define T_SZ 2048
#define E_SZ 1024
#define NH   16
#define HD   64

#define KT        64   // GEMM K-tile (2 WMMA K-steps per LDS fill)
#define LDS_STRIDE 80  // 64 data elements + 16 pad -> bank-friendly, 16B-aligned rows

#if defined(__has_builtin)
#if __has_builtin(__builtin_amdgcn_tensor_load_to_lds)
#define USE_TDM 1
#endif
#endif

__device__ __forceinline__ unsigned short f32_to_bf16_bits(float f) {
  union { float f; unsigned u; } v; v.f = f;
  unsigned x = v.u;
  if ((x & 0x7fffffffu) > 0x7f800000u) return (unsigned short)((x >> 16) | 0x0040u); // quiet NaN
  return (unsigned short)((x + 0x7fffu + ((x >> 16) & 1u)) >> 16);                   // RNE
}
__device__ __forceinline__ bf16_t f32_to_bf16(float f) {
  unsigned short b = f32_to_bf16_bits(f);
  return __builtin_bit_cast(bf16_t, b);
}
// for values known finite (exp outputs, normalized accums)
__device__ __forceinline__ bf16_t f32_to_bf16_fast(float f) {
  union { float f; unsigned u; } v; v.f = f;
  unsigned short b = (unsigned short)((v.u + 0x7fffu + ((v.u >> 16) & 1u)) >> 16);
  return __builtin_bit_cast(bf16_t, b);
}

union FragBF {
  v16bf v;
  unsigned u[8];
  unsigned short h[16];
  uint4 q[2];
};

#if USE_TDM
typedef unsigned int v4u_t __attribute__((ext_vector_type(4)));
typedef int v8i_t __attribute__((ext_vector_type(8)));
typedef int v4i_t __attribute__((ext_vector_type(4)));

// 2D bf16 tile load via Tensor Data Mover. tile_k elems/row, tile_rows rows,
// global row stride row_stride elems. LDS rows padded: pad_interval_code selects
// data run (3 = 16 DW = 32 elems ; 4 = 32 DW = 64 elems), pad_amount = 8 DW (16 elems).
__device__ __forceinline__ void tdm_load_2d(const bf16_t* gptr, unsigned lds_off,
                                            unsigned tile_k, unsigned tile_rows,
                                            unsigned row_stride,
                                            unsigned pad_interval_code) {
  unsigned long long ga = (unsigned long long)(uintptr_t)gptr;
  v4u_t g0;
  g0[0] = 1u;                                                  // count=1
  g0[1] = lds_off;                                             // lds_addr (bytes)
  g0[2] = (unsigned)ga;                                        // global_addr lo
  g0[3] = (unsigned)((ga >> 32) & 0x01FFFFFFu) | 0x80000000u;  // addr[56:32] | type=2
  v8i_t g1;
  // data_size=1 (2B) | pad_enable | pad_interval | pad_amount=7 (8 DW)
  g1[0] = (int)((1u << 16) | (1u << 20) | (pad_interval_code << 22) | (7u << 25));
  const unsigned dim0 = 0x40000000u, dim1 = 0x40000000u;       // huge: no OOB clip
  g1[1] = (int)((dim0 & 0xFFFFu) << 16);                       // tensor_dim0 lo16
  g1[2] = (int)(((dim0 >> 16) & 0xFFFFu) | ((dim1 & 0xFFFFu) << 16));
  g1[3] = (int)(((dim1 >> 16) & 0xFFFFu) | (tile_k << 16));    // tile_dim0
  g1[4] = (int)(tile_rows & 0xFFFFu);                          // tile_dim1
  g1[5] = (int)row_stride;                                     // tensor_dim0_stride lo32
  g1[6] = 0;
  g1[7] = 0;
  v4i_t z4 = {0, 0, 0, 0};
  v8i_t z8 = {0, 0, 0, 0, 0, 0, 0, 0};
  __builtin_amdgcn_tensor_load_to_lds(g0, g1, z4, z4, z8, 0);
}

__device__ __forceinline__ void tdm_wait0() {
#if __has_builtin(__builtin_amdgcn_s_wait_tensorcnt)
  __builtin_amdgcn_s_wait_tensorcnt(0);
#else
  asm volatile("s_wait_tensorcnt 0" ::: "memory");
#endif
}
__device__ __forceinline__ void tdm_wait2() {
#if __has_builtin(__builtin_amdgcn_s_wait_tensorcnt)
  __builtin_amdgcn_s_wait_tensorcnt(2);
#else
  asm volatile("s_wait_tensorcnt 2" ::: "memory");
#endif
}
#endif

// ---------------------------------------------------------------- converts
__global__ void cvt_f32_bf16(const float* __restrict__ in, bf16_t* __restrict__ out, int n) {
  int i = blockIdx.x * blockDim.x + threadIdx.x;
  int stride = gridDim.x * blockDim.x;
  for (; i < n; i += stride) out[i] = f32_to_bf16(in[i]);
}

// f32 [K][N] -> bf16 [N][K] (weights, done once; makes GEMM B-tiles contiguous)
__global__ __launch_bounds__(256)
void cvt_transpose_f32_bf16(const float* __restrict__ in, bf16_t* __restrict__ out,
                            int K, int N) {
  __shared__ bf16_t Ts[64][72];
  const int k0 = blockIdx.x * 64, n0 = blockIdx.y * 64;
  const int tid = threadIdx.x;
  {
    const int k = tid >> 2, c = (tid & 3) * 16;
    const float* src = in + (size_t)(k0 + k) * N + n0 + c;
#pragma unroll
    for (int i = 0; i < 16; ++i) Ts[k][c + i] = f32_to_bf16(src[i]);
  }
  __syncthreads();
  {
    const int n = tid >> 2, kb = (tid & 3) * 16;
    alignas(16) unsigned short tmp[16];
#pragma unroll
    for (int i = 0; i < 16; ++i)
      tmp[i] = __builtin_bit_cast(unsigned short, Ts[kb + i][n]);
    uint4* dst = (uint4*)(out + (size_t)(n0 + n) * K + k0 + kb);
    dst[0] = *(uint4*)&tmp[0];
    dst[1] = *(uint4*)&tmp[8];
  }
}

// V slice of kqv [B*T,3E] -> Vt [B,H,D,T] so attention B-fragments of V are contiguous
__global__ __launch_bounds__(256)
void transpose_v(const bf16_t* __restrict__ kqv, bf16_t* __restrict__ Vt) {
  __shared__ bf16_t Ts[64][72];
  const int t0 = blockIdx.x * 64;
  const int bh = blockIdx.y;  // b*NH + h
  const int b = bh >> 4, h = bh & 15;
  const int tid = threadIdx.x;
  {
    const int t = tid >> 2, c = (tid & 3) * 16;
    const bf16_t* src = kqv + (size_t)(b * T_SZ + t0 + t) * (3 * E_SZ) + 2 * E_SZ + h * HD + c;
    *(uint4*)&Ts[t][c]     = *(const uint4*)src;
    *(uint4*)&Ts[t][c + 8] = *(const uint4*)(src + 8);
  }
  __syncthreads();
  {
    const int d = tid >> 2, tb = (tid & 3) * 16;
    alignas(16) unsigned short tmp[16];
#pragma unroll
    for (int i = 0; i < 16; ++i)
      tmp[i] = __builtin_bit_cast(unsigned short, Ts[tb + i][d]);
    uint4* dst = (uint4*)(Vt + ((size_t)bh * HD + d) * T_SZ + t0 + tb);
    dst[0] = *(uint4*)&tmp[0];
    dst[1] = *(uint4*)&tmp[8];
  }
}

// ---------------------------------------------------------------- GEMM
// C[M,N] = A[M,K](bf16 row-major) * B[K,N], B supplied TRANSPOSED: Bt[N][K].
// Block 128x128 (8 waves), wave tile 32x64 (2x4 WMMA), K-tile 64 (2 K-steps).
// Double-buffered TDM: DMA of tile i+1 overlaps WMMA on tile i.
__global__ __launch_bounds__(256)
void gemm_bf16_wmma(const bf16_t* __restrict__ A, const bf16_t* __restrict__ Bt,
                    const float* __restrict__ bias,
                    bf16_t* __restrict__ Cb, float* __restrict__ Cf,
                    int M, int N, int K) {
  __shared__ bf16_t As[2][128 * LDS_STRIDE];  // 2 x 20 KB
  __shared__ bf16_t Bs[2][128 * LDS_STRIDE];  // 2 x 20 KB

  const int tid  = threadIdx.x;
  const int lane = tid & 31, wave = tid >> 5;
  const int half = lane >> 4, l16 = lane & 15;
  const int wm = (wave & 3) * 32, wn = (wave >> 2) * 64;
  const int bm = blockIdx.x * 128, bn = blockIdx.y * 128;

  v8f acc[2][4];
#pragma unroll
  for (int mi = 0; mi < 2; ++mi)
#pragma unroll
    for (int ni = 0; ni < 4; ++ni)
#pragma unroll
      for (int j = 0; j < 8; ++j) acc[mi][ni][j] = 0.0f;

  const int nk = K / KT;

#if USE_TDM
  if (wave == 0) {  // prologue: fill buffer 0
    tdm_load_2d(A  + (size_t)bm * K, (unsigned)(uintptr_t)&As[0][0], KT, 128u, (unsigned)K, 4u);
    tdm_load_2d(Bt + (size_t)bn * K, (unsigned)(uintptr_t)&Bs[0][0], KT, 128u, (unsigned)K, 4u);
  }
#endif

  int buf = 0;
  for (int ki = 0; ki < nk; ++ki, buf ^= 1) {
#if USE_TDM
    if (wave == 0) {
      if (ki + 1 < nk) {  // kick DMA for next tile into the other buffer
        const int k1 = (ki + 1) * KT;
        tdm_load_2d(A  + (size_t)bm * K + k1, (unsigned)(uintptr_t)&As[buf ^ 1][0],
                    KT, 128u, (unsigned)K, 4u);
        tdm_load_2d(Bt + (size_t)bn * K + k1, (unsigned)(uintptr_t)&Bs[buf ^ 1][0],
                    KT, 128u, (unsigned)K, 4u);
        tdm_wait2();  // in-order drain: <=2 outstanding => current buffer resident
      } else {
        tdm_wait0();
      }
    }
    __syncthreads();
#else
    {
      const int fr = tid >> 1, fc = (tid & 1) * 32;
      const bf16_t* ag = A + (size_t)(bm + fr) * K + ki * KT + fc;
      const bf16_t* bg = Bt + (size_t)(bn + fr) * K + ki * KT + fc;
      bf16_t* al = &As[buf][fr * LDS_STRIDE + fc];
      bf16_t* bl = &Bs[buf][fr * LDS_STRIDE + fc];
#pragma unroll
      for (int i = 0; i < 4; ++i) {
        *(uint4*)(al + i * 8) = *(const uint4*)(ag + i * 8);
        *(uint4*)(bl + i * 8) = *(const uint4*)(bg + i * 8);
      }
    }
    __syncthreads();
#endif

    const bf16_t* as = &As[buf][0];
    const bf16_t* bs = &Bs[buf][0];
#pragma unroll
    for (int s = 0; s < 2; ++s) {  // two WMMA K-steps per tile
      // A fragment (16x32, A-layout): q[0]=k 8h.. ; q[1]=k 16+8h..
      FragBF af[2];
#pragma unroll
      for (int mi = 0; mi < 2; ++mi) {
        const bf16_t* ap = as + (wm + mi * 16 + l16) * LDS_STRIDE + s * 32 + 8 * half;
        af[mi].q[0] = *(const uint4*)ap;
        af[mi].q[1] = *(const uint4*)(ap + 16);
      }
      // B fragment (32x16, B-layout): q[0]=k 8h.. ; q[1]=k 16+8h..
      FragBF bfr[4];
#pragma unroll
      for (int ni = 0; ni < 4; ++ni) {
        const bf16_t* bp = bs + (wn + ni * 16 + l16) * LDS_STRIDE + s * 32 + 8 * half;
        bfr[ni].q[0] = *(const uint4*)bp;
        bfr[ni].q[1] = *(const uint4*)(bp + 16);
      }
#pragma unroll
      for (int mi = 0; mi < 2; ++mi)
#pragma unroll
        for (int ni = 0; ni < 4; ++ni)
          acc[mi][ni] = __builtin_amdgcn_wmma_f32_16x16x32_bf16(
              false, af[mi].v, false, bfr[ni].v, (short)0, acc[mi][ni], false, false);
    }
    __syncthreads();  // all reads of buf done before its next DMA overwrite
  }

  // epilogue: C layout element j -> row j + 8*half, col l16
#pragma unroll
  for (int mi = 0; mi < 2; ++mi)
#pragma unroll
    for (int ni = 0; ni < 4; ++ni)
#pragma unroll
      for (int j = 0; j < 8; ++j) {
        const int r = bm + wm + mi * 16 + j + 8 * half;
        const int c = bn + wn + ni * 16 + l16;
        float vv = acc[mi][ni][j];
        if (Cf) Cf[(size_t)r * N + c] = vv + (bias ? bias[c] : 0.0f);
        else    Cb[(size_t)r * N + c] = f32_to_bf16_fast(vv);
      }
}

// ---------------------------------------------------------------- attention
// kqv: [B*T,3E] (k|q|v), Vt: [B,H,D,T]. One wave = 16 query rows, flash-style
// online softmax over 32-key blocks; Q resident, K/V fragments are contiguous
// global b128 loads, P re-layout via per-wave LDS.
__global__ __launch_bounds__(256)
void attn_wmma(const bf16_t* __restrict__ kqv, const bf16_t* __restrict__ Vt,
               bf16_t* __restrict__ xatt) {
  __shared__ bf16_t Ps[8][16 * 32];  // per-wave P buffer (8 KB total)

  const int tid  = threadIdx.x;
  const int lane = tid & 31, wave = tid >> 5;
  const int half = lane >> 4, l16 = lane & 15;

  const int bh = blockIdx.y;
  const int b = bh >> 4, h = bh & 15;
  const int q0 = blockIdx.x * 128 + wave * 16;

  const size_t rs = 3 * E_SZ;
  const bf16_t* base  = kqv + (size_t)b * T_SZ * rs;
  const bf16_t* vbase = Vt + (size_t)bh * HD * T_SZ;

  // Q fragments (A-layout), resident: 16 rows x 64 d = 2 k-steps
  FragBF qf[2];
  {
    const bf16_t* qrow = base + (size_t)(q0 + l16) * rs + E_SZ + h * HD;
#pragma unroll
    for (int st = 0; st < 2; ++st) {
      qf[st].q[0] = *(const uint4*)(qrow + st * 32 + 8 * half);
      qf[st].q[1] = *(const uint4*)(qrow + st * 32 + 16 + 8 * half);
    }
  }

  v8f o[4];
  float m_run[8], l_run[8];
#pragma unroll
  for (int dt = 0; dt < 4; ++dt)
#pragma unroll
    for (int j = 0; j < 8; ++j) o[dt][j] = 0.0f;
#pragma unroll
  for (int j = 0; j < 8; ++j) { m_run[j] = -1e30f; l_run[j] = 0.0f; }

  const int kend = q0 + 16;  // causal bound, wave-uniform
  for (int kb = 0; kb < kend; kb += 32) {
    // ---- S = Q K^T : K^T B-fragments = contiguous runs of K rows
    v8f s[2];
#pragma unroll
    for (int nt = 0; nt < 2; ++nt) {
#pragma unroll
      for (int j = 0; j < 8; ++j) s[nt][j] = 0.0f;
      const bf16_t* krow = base + (size_t)(kb + nt * 16 + l16) * rs + h * HD;
#pragma unroll
      for (int st = 0; st < 2; ++st) {
        FragBF kf;
        const bf16_t* kp = krow + st * 32 + 8 * half;
        kf.q[0] = *(const uint4*)kp;
        kf.q[1] = *(const uint4*)(kp + 16);
        s[nt] = __builtin_amdgcn_wmma_f32_16x16x32_bf16(
            false, qf[st].v, false, kf.v, (short)0, s[nt], false, false);
      }
    }

    // ---- online softmax (row m = j + 8*half lives across the 16-lane half)
    float pv[2][8], mnew[8];
#pragma unroll
    for (int j = 0; j < 8; ++j) {
      const int qg = q0 + j + 8 * half;
      float v0 = s[0][j] * 0.125f;  // HEAD^-0.5
      float v1 = s[1][j] * 0.125f;
      if (kb + l16 > qg)      v0 = -1e30f;
      if (kb + 16 + l16 > qg) v1 = -1e30f;
      pv[0][j] = v0; pv[1][j] = v1;
      float mx = fmaxf(v0, v1);
#pragma unroll
      for (int d = 1; d < 16; d <<= 1) mx = fmaxf(mx, __shfl_xor(mx, d, 32));
      mnew[j] = fmaxf(m_run[j], mx);
    }
#pragma unroll
    for (int j = 0; j < 8; ++j) {
      const float corr = __expf(m_run[j] - mnew[j]);
      const float p0 = __expf(pv[0][j] - mnew[j]);
      const float p1 = __expf(pv[1][j] - mnew[j]);
      pv[0][j] = p0; pv[1][j] = p1;
      float rsum = p0 + p1;
#pragma unroll
      for (int d = 1; d < 16; d <<= 1) rsum += __shfl_xor(rsum, d, 32);
      l_run[j] = l_run[j] * corr + rsum;
      m_run[j] = mnew[j];
#pragma unroll
      for (int dt = 0; dt < 4; ++dt) o[dt][j] *= corr;
    }

    // ---- P (C layout) -> LDS row-major -> A fragment
    bf16_t* pl = &Ps[wave][0];
#pragma unroll
    for (int nt = 0; nt < 2; ++nt)
#pragma unroll
      for (int j = 0; j < 8; ++j)
        pl[(j + 8 * half) * 32 + nt * 16 + l16] = f32_to_bf16_fast(pv[nt][j]);
    __builtin_amdgcn_wave_barrier();
    asm volatile("s_wait_dscnt 0" ::: "memory");  // same-wave LDS RAW fence

    FragBF pf;
    pf.q[0] = *(const uint4*)&pl[l16 * 32 + 8 * half];
    pf.q[1] = *(const uint4*)&pl[l16 * 32 + 16 + 8 * half];

    // ---- O += P @ V : V B-fragments contiguous from Vt [B,H,D,T]
#pragma unroll
    for (int dt = 0; dt < 4; ++dt) {
      FragBF vf;
      const bf16_t* vp = vbase + (size_t)(dt * 16 + l16) * T_SZ + kb + 8 * half;
      vf.q[0] = *(const uint4*)vp;
      vf.q[1] = *(const uint4*)(vp + 16);
      o[dt] = __builtin_amdgcn_wmma_f32_16x16x32_bf16(
          false, pf.v, false, vf.v, (short)0, o[dt], false, false);
    }
    __builtin_amdgcn_wave_barrier();
  }

  // ---- normalize + write x_att[b][t][h*64+d] (bf16 feeds proj GEMM)
#pragma unroll
  for (int j = 0; j < 8; ++j) {
    const int qg = q0 + j + 8 * half;
    const float inv = 1.0f / l_run[j];
#pragma unroll
    for (int dt = 0; dt < 4; ++dt)
      xatt[(size_t)(b * T_SZ + qg) * E_SZ + h * HD + dt * 16 + l16] =
          f32_to_bf16_fast(o[dt][j] * inv);
  }
}

// ---------------------------------------------------------------- launch
extern "C" void kernel_launch(void* const* d_in, const int* in_sizes, int n_in,
                              void* d_out, int out_size, void* d_ws, size_t ws_size,
                              hipStream_t stream) {
  (void)in_sizes; (void)n_in; (void)out_size; (void)ws_size;
  const float* x      = (const float*)d_in[0];
  const float* W_kqv  = (const float*)d_in[1];
  const float* W_proj = (const float*)d_in[2];
  const float* b_proj = (const float*)d_in[3];
  float* out = (float*)d_out;

  const int M = B_SZ * T_SZ;  // 8192

  // workspace carve (bf16): 104 MB
  bf16_t* ws      = (bf16_t*)d_ws;
  bf16_t* xb      = ws;                                  // M*E
  bf16_t* wkqvT   = xb     + (size_t)M * E_SZ;           // [3E][E]
  bf16_t* wprojT  = wkqvT  + (size_t)3 * E_SZ * E_SZ;    // [E][E]
  bf16_t* kqvb    = wprojT + (size_t)E_SZ * E_SZ;        // M*3E
  bf16_t* xattb   = kqvb   + (size_t)M * 3 * E_SZ;       // M*E
  bf16_t* vtb     = xattb  + (size_t)M * E_SZ;           // [B,H,D,T] = M*E

  cvt_f32_bf16<<<2048, 256, 0, stream>>>(x, xb, M * E_SZ);
  cvt_transpose_f32_bf16<<<dim3(E_SZ / 64, (3 * E_SZ) / 64), 256, 0, stream>>>(
      W_kqv, wkqvT, E_SZ, 3 * E_SZ);
  cvt_transpose_f32_bf16<<<dim3(E_SZ / 64, E_SZ / 64), 256, 0, stream>>>(
      W_proj, wprojT, E_SZ, E_SZ);

  // kqv = x @ W_kqv   [8192 x 3072]
  gemm_bf16_wmma<<<dim3(M / 128, (3 * E_SZ) / 128), 256, 0, stream>>>(
      xb, wkqvT, nullptr, kqvb, nullptr, M, 3 * E_SZ, E_SZ);

  // V -> [B,H,D,T] for contiguous attention fragments
  transpose_v<<<dim3(T_SZ / 64, B_SZ * NH), 256, 0, stream>>>(kqvb, vtb);

  // fused causal attention -> x_att [8192 x 1024]
  attn_wmma<<<dim3(T_SZ / 128, B_SZ * NH), 256, 0, stream>>>(kqvb, vtb, xattb);

  // out = x_att @ W_proj + b_proj  (f32)
  gemm_bf16_wmma<<<dim3(M / 128, E_SZ / 128), 256, 0, stream>>>(
      xattb, wprojT, b_proj, nullptr, out, M, E_SZ, E_SZ);
}
